// SpUNetBase_67439576482090
// MI455X (gfx1250) — matrix operand order repaired and checked by reading, not compiled, not run on hardware
//
#include <hip/hip_runtime.h>

// ---------------------------------------------------------------------------
// Types for CDNA5 WMMA (wave32): v16bf A/B fragments, v8f accumulators.
// ---------------------------------------------------------------------------
typedef __bf16 bf16_t;
typedef __attribute__((ext_vector_type(16))) __bf16        v16bf;
typedef __attribute__((ext_vector_type(8)))  float         v8f;
typedef __attribute__((ext_vector_type(4)))  unsigned int  u32x4;

union FragBF { v16bf v; u32x4 u[2]; };

#define NVOX  200000
#define CCH   96
#define KOFF  27
#define PBANK 256
#define BN_EPS 1e-3f
#define LN_EPS 1e-5f
#define INV_SQRT_MID 0.10206207261596575f   // 1/sqrt(96)

// ---------------------------------------------------------------------------
// Prep kernels
// ---------------------------------------------------------------------------
__global__ void zero_stats_kernel(float* stats) {
    int i = threadIdx.x;
    if (i < 4 * CCH) stats[i] = 0.0f;
}

// W (K,Cin,Cout) fp32 -> Wt (K,Cout,Cin) bf16   [B-fragment friendly: [N][K] contiguous]
__global__ void prep_w_kernel(const float* __restrict__ W, bf16_t* __restrict__ Wt, int total) {
    int i = blockIdx.x * blockDim.x + threadIdx.x;
    if (i >= total) return;
    int k  = i / (CCH * CCH);
    int r  = i % (CCH * CCH);
    int co = r / CCH;
    int ci = r % CCH;
    Wt[i] = (bf16_t)W[(k * CCH + ci) * CCH + co];
}

// Wk (C,MID) fp32 -> WkT (MID,C) bf16
__global__ void prep_wk_kernel(const float* __restrict__ Wk, bf16_t* __restrict__ WkT) {
    int i = blockIdx.x * blockDim.x + threadIdx.x;
    if (i >= CCH * CCH) return;
    int m = i / CCH, c = i % CCH;
    WkT[m * CCH + c] = (bf16_t)Wk[c * CCH + m];
}

// fp32 -> bf16 elementwise
__global__ void prep_feat_kernel(const float* __restrict__ x, bf16_t* __restrict__ y, int total) {
    int i = blockIdx.x * blockDim.x + threadIdx.x;
    if (i >= total) return;
    y[i] = (bf16_t)x[i];
}

// q = pf@Wq + bq (scaled by 1/sqrt(MID)) -> qB bf16 [P][MID]
// vT[o][p] = (pf[p]@Wv[:,o] + bv[o])     -> vT bf16 [OUT][P]
__global__ void prep_qv_kernel(const float* __restrict__ pf,
                               const float* __restrict__ Wq, const float* __restrict__ bq,
                               const float* __restrict__ Wv, const float* __restrict__ bv,
                               bf16_t* __restrict__ qB, bf16_t* __restrict__ vT) {
    int i = blockIdx.x * blockDim.x + threadIdx.x;
    if (i >= PBANK * CCH) return;
    int p = i / CCH, j = i % CCH;
    float sq = bq[j], sv = bv[j];
    for (int c = 0; c < CCH; ++c) {
        float f = pf[p * CCH + c];
        sq += f * Wq[c * CCH + j];
        sv += f * Wv[c * CCH + j];
    }
    qB[p * CCH + j]   = (bf16_t)(sq * INV_SQRT_MID);
    vT[j * PBANK + p] = (bf16_t)sv;
}

// ---------------------------------------------------------------------------
// Submanifold conv via rulebook gather + WMMA:
// out[N,96] fp32 = sum_k inB[nbr[:,k]] @ W[k]    (Wt layout [k][cout][cin] bf16)
// One wave = 32 rows x 96 cols (two 16-row A tiles share each B fragment ->
// 1.33 b128 loads per WMMA). 8 waves / 256-thread block = 256 rows / block.
// Rulebook indices are software-pipelined one k ahead + global_prefetch of the
// next gathered rows (rows are L2-resident: whole bf16 feature table = 38 MB).
// ---------------------------------------------------------------------------
__global__ __launch_bounds__(256)
void conv_wmma_kernel(const bf16_t* __restrict__ inB, const bf16_t* __restrict__ Wt,
                      const int* __restrict__ nbr, float* __restrict__ outF, int nRows) {
    const int lane = threadIdx.x & 31;
    const int wave = threadIdx.x >> 5;
    const int half = lane >> 4;
    const int lm   = lane & 15;
    const int tile0 = (blockIdx.x * 8 + wave) * 32;
    if (tile0 >= nRows) return;                 // uniform per wave (nRows % 32 == 0)
    const int row0 = tile0 + lm;
    const int row1 = tile0 + 16 + lm;

    v8f acc[2][6];
#pragma unroll
    for (int t = 0; t < 2; ++t)
#pragma unroll
        for (int i = 0; i < 6; ++i) acc[t][i] = (v8f){0.f,0.f,0.f,0.f,0.f,0.f,0.f,0.f};

    const int kb0 = half * 8;    // A-fragment K base within 32-chunk
    const int kbB = half * 8;    // B-fragment K base within 32-chunk

    int g0 = nbr[row0 * KOFF];
    int g1 = nbr[row1 * KOFF];
    for (int k = 0; k < KOFF; ++k) {
        const bf16_t* arow0 = inB + (long)g0 * CCH;
        const bf16_t* arow1 = inB + (long)g1 * CCH;
        if (k + 1 < KOFF) {                     // pipeline next rulebook entry
            g0 = nbr[row0 * KOFF + k + 1];
            g1 = nbr[row1 * KOFF + k + 1];
            __builtin_prefetch(inB + (long)g0 * CCH, 0, 3);
            __builtin_prefetch(inB + (long)g1 * CCH, 0, 3);
        }
        const bf16_t* wk = Wt + (long)k * CCH * CCH;
#pragma unroll
        for (int ch = 0; ch < 3; ++ch) {
            FragBF a0, a1;
            const bf16_t* ap0 = arow0 + ch * 32 + kb0;
            const bf16_t* ap1 = arow1 + ch * 32 + kb0;
            a0.u[0] = *(const u32x4*)(ap0);
            a0.u[1] = *(const u32x4*)(ap0 + 16);
            a1.u[0] = *(const u32x4*)(ap1);
            a1.u[1] = *(const u32x4*)(ap1 + 16);
            const bf16_t* wbase = wk + ch * 32 + kbB;
#pragma unroll
            for (int nt = 0; nt < 6; ++nt) {
                FragBF b;
                const bf16_t* bp = wbase + (nt * 16 + lm) * CCH;
                b.u[0] = *(const u32x4*)(bp);
                b.u[1] = *(const u32x4*)(bp + 16);
                acc[0][nt] = __builtin_amdgcn_wmma_f32_16x16x32_bf16(
                    false, a0.v, false, b.v, (short)0, acc[0][nt], false, false);
                acc[1][nt] = __builtin_amdgcn_wmma_f32_16x16x32_bf16(
                    false, a1.v, false, b.v, (short)0, acc[1][nt], false, false);
            }
        }
    }
#pragma unroll
    for (int t = 0; t < 2; ++t)
#pragma unroll
        for (int nt = 0; nt < 6; ++nt)
#pragma unroll
            for (int r = 0; r < 8; ++r) {
                long m = tile0 + t * 16 + r + half * 8;
                outF[m * CCH + nt * 16 + lm] = acc[t][nt][r];
            }
}

// ---------------------------------------------------------------------------
// Batch-norm statistics: per-channel sum / sumsq (96 threads own 96 channels)
// ---------------------------------------------------------------------------
__global__ __launch_bounds__(96)
void bn_stats_kernel(const float* __restrict__ x, float* __restrict__ stats,
                     int nRows, int rowsPerBlock) {
    int c = threadIdx.x;
    long r0 = (long)blockIdx.x * rowsPerBlock;
    long r1 = r0 + rowsPerBlock; if (r1 > nRows) r1 = nRows;
    float s = 0.f, q = 0.f;
    for (long r = r0; r < r1; ++r) {
        float v = x[r * CCH + c];
        s += v; q += v * v;
    }
    atomicAdd(&stats[c], s);
    atomicAdd(&stats[CCH + c], q);
}

// BN1 apply + ReLU -> bf16 (input to conv2)
__global__ void bn1_apply_kernel(const float* __restrict__ x, const float* __restrict__ stats,
                                 const float* __restrict__ gamma, const float* __restrict__ beta,
                                 const int* __restrict__ cond, bf16_t* __restrict__ y, int total) {
    int i = blockIdx.x * blockDim.x + threadIdx.x;
    if (i >= total) return;
    int c = i % CCH;
    int cd = cond[0];
    float mu  = stats[c] * (1.0f / NVOX);
    float var = stats[CCH + c] * (1.0f / NVOX) - mu * mu;
    float xn  = (x[i] - mu) * rsqrtf(var + BN_EPS);
    float v   = xn * gamma[cd * CCH + c] + beta[cd * CCH + c];
    y[i] = (bf16_t)fmaxf(v, 0.0f);
}

// BN2 apply + residual + ReLU -> fp32 output cols [0,96) and bf16 copy for attention
__global__ void bn2_apply_kernel(const float* __restrict__ x, const float* __restrict__ stats,
                                 const float* __restrict__ gamma, const float* __restrict__ beta,
                                 const int* __restrict__ cond, const float* __restrict__ resid,
                                 float* __restrict__ outFull, bf16_t* __restrict__ outB, int total) {
    int i = blockIdx.x * blockDim.x + threadIdx.x;
    if (i >= total) return;
    int c = i % CCH;
    long r = i / CCH;
    int cd = cond[0];
    float mu  = stats[c] * (1.0f / NVOX);
    float var = stats[CCH + c] * (1.0f / NVOX) - mu * mu;
    float xn  = (x[i] - mu) * rsqrtf(var + BN_EPS);
    float v   = xn * gamma[cd * CCH + c] + beta[cd * CCH + c] + resid[i];
    v = fmaxf(v, 0.0f);
    outFull[r * (2 * CCH) + c] = v;
    outB[i] = (bf16_t)v;
}

// ---------------------------------------------------------------------------
// Fused cross-attention + LayerNorm, one wave per 16 voxels, 4 waves/block.
//   kvec = out @ WkT^T + bk           (18 WMMAs)
//   logits = kvec @ q^T (prescaled)   (48 WMMAs, 16x256 in registers)
//   softmax over 256 (shfl_xor within 16-lane halves)
//   att = w @ vT^T                    (48 WMMAs)
//   LayerNorm over 96 -> out cols [96,192)
// ---------------------------------------------------------------------------
__global__ __launch_bounds__(128)
void attn_wmma_kernel(const bf16_t* __restrict__ outB, const bf16_t* __restrict__ WkT,
                      const float* __restrict__ bk, const bf16_t* __restrict__ qB,
                      const bf16_t* __restrict__ vT, float* __restrict__ outFull, int nRows) {
    __shared__ __align__(16) bf16_t lds[4 * 16 * PBANK];   // 32 KB, 16x256 bf16 per wave
    const int lane = threadIdx.x & 31;
    const int wave = threadIdx.x >> 5;
    const int half = lane >> 4;
    const int lm   = lane & 15;
    const long tile0 = ((long)blockIdx.x * 4 + wave) * 16;
    bf16_t* wsh = lds + wave * 16 * PBANK;

    const int kb0 = half * 8;
    const int kbB = half * 8;

    // ---- 1) k-projection: kvec = outB_tile @ Wk + bk -----------------------
    v8f kacc[6];
#pragma unroll
    for (int i = 0; i < 6; ++i) kacc[i] = (v8f){0.f,0.f,0.f,0.f,0.f,0.f,0.f,0.f};
    const bf16_t* arow = outB + (tile0 + lm) * CCH;
#pragma unroll
    for (int ch = 0; ch < 3; ++ch) {
        FragBF a;
        const bf16_t* ap = arow + ch * 32 + kb0;
        a.u[0] = *(const u32x4*)(ap);
        a.u[1] = *(const u32x4*)(ap + 16);
#pragma unroll
        for (int nt = 0; nt < 6; ++nt) {
            FragBF b;
            const bf16_t* bp = WkT + (nt * 16 + lm) * CCH + ch * 32 + kbB;
            b.u[0] = *(const u32x4*)(bp);
            b.u[1] = *(const u32x4*)(bp + 16);
            kacc[nt] = __builtin_amdgcn_wmma_f32_16x16x32_bf16(
                false, a.v, false, b.v, (short)0, kacc[nt], false, false);
        }
    }
    // + bk, stage as [M][96] bf16 in LDS for the next GEMM's A fragments
#pragma unroll
    for (int nt = 0; nt < 6; ++nt)
#pragma unroll
        for (int r = 0; r < 8; ++r) {
            int M = r + half * 8;
            int chn = nt * 16 + lm;
            wsh[M * CCH + chn] = (bf16_t)(kacc[nt][r] + bk[chn]);
        }
    __syncthreads();

    // ---- 2) logits = kvec @ q^T (q prescaled by 1/sqrt(MID)) ---------------
    v8f l[16];
#pragma unroll
    for (int i = 0; i < 16; ++i) l[i] = (v8f){0.f,0.f,0.f,0.f,0.f,0.f,0.f,0.f};
#pragma unroll
    for (int ch = 0; ch < 3; ++ch) {
        FragBF a;
        const bf16_t* ap = wsh + lm * CCH + ch * 32 + kb0;
        a.u[0] = *(const u32x4*)(ap);
        a.u[1] = *(const u32x4*)(ap + 16);
#pragma unroll
        for (int nt = 0; nt < 16; ++nt) {
            FragBF b;
            const bf16_t* bp = qB + (nt * 16 + lm) * CCH + ch * 32 + kbB;
            b.u[0] = *(const u32x4*)(bp);
            b.u[1] = *(const u32x4*)(bp + 16);
            l[nt] = __builtin_amdgcn_wmma_f32_16x16x32_bf16(
                false, a.v, false, b.v, (short)0, l[nt], false, false);
        }
    }
    __syncthreads();

    // ---- 3) softmax over 256 points per row --------------------------------
#pragma unroll
    for (int r = 0; r < 8; ++r) {
        float mx = -3.402823466e38f;
#pragma unroll
        for (int nt = 0; nt < 16; ++nt) mx = fmaxf(mx, l[nt][r]);
#pragma unroll
        for (int m = 1; m < 16; m <<= 1) mx = fmaxf(mx, __shfl_xor(mx, m, 32));
        float s = 0.f;
#pragma unroll
        for (int nt = 0; nt < 16; ++nt) {
            float e = __expf(l[nt][r] - mx);
            l[nt][r] = e;
            s += e;
        }
#pragma unroll
        for (int m = 1; m < 16; m <<= 1) s += __shfl_xor(s, m, 32);
        float inv = 1.0f / s;
        int M = r + half * 8;
#pragma unroll
        for (int nt = 0; nt < 16; ++nt)
            wsh[M * PBANK + nt * 16 + lm] = (bf16_t)(l[nt][r] * inv);
    }
    __syncthreads();

    // ---- 4) att = w @ v   (K = 256, 8 chunks) ------------------------------
    v8f aacc[6];
#pragma unroll
    for (int i = 0; i < 6; ++i) aacc[i] = (v8f){0.f,0.f,0.f,0.f,0.f,0.f,0.f,0.f};
#pragma unroll
    for (int ch = 0; ch < 8; ++ch) {
        FragBF a;
        const bf16_t* ap = wsh + lm * PBANK + ch * 32 + kb0;
        a.u[0] = *(const u32x4*)(ap);
        a.u[1] = *(const u32x4*)(ap + 16);
#pragma unroll
        for (int nt = 0; nt < 6; ++nt) {
            FragBF b;
            const bf16_t* bp = vT + (nt * 16 + lm) * PBANK + ch * 32 + kbB;
            b.u[0] = *(const u32x4*)(bp);
            b.u[1] = *(const u32x4*)(bp + 16);
            aacc[nt] = __builtin_amdgcn_wmma_f32_16x16x32_bf16(
                false, a.v, false, b.v, (short)0, aacc[nt], false, false);
        }
    }

    // ---- 5) LayerNorm over 96 channels per row, store cols [96,192) --------
#pragma unroll
    for (int r = 0; r < 8; ++r) {
        float s = 0.f, q2 = 0.f;
#pragma unroll
        for (int nt = 0; nt < 6; ++nt) { float x = aacc[nt][r]; s += x; q2 += x * x; }
#pragma unroll
        for (int m = 1; m < 16; m <<= 1) { s += __shfl_xor(s, m, 32); q2 += __shfl_xor(q2, m, 32); }
        float mu  = s * (1.0f / CCH);
        float var = q2 * (1.0f / CCH) - mu * mu;
        float inv = rsqrtf(var + LN_EPS);
        int M = r + half * 8;
#pragma unroll
        for (int nt = 0; nt < 6; ++nt)
            outFull[(tile0 + M) * (2 * CCH) + CCH + nt * 16 + lm] = (aacc[nt][r] - mu) * inv;
    }
}

// ---------------------------------------------------------------------------
// Host launcher
// ---------------------------------------------------------------------------
extern "C" void kernel_launch(void* const* d_in, const int* in_sizes, int n_in,
                              void* d_out, int out_size, void* d_ws, size_t ws_size,
                              hipStream_t stream) {
    const float* voxel_feat = (const float*)d_in[0];
    const float* point_feat = (const float*)d_in[1];
    const float* W1         = (const float*)d_in[2];
    const float* W2         = (const float*)d_in[3];
    const float* gamma1     = (const float*)d_in[4];
    const float* beta1      = (const float*)d_in[5];
    const float* gamma2     = (const float*)d_in[6];
    const float* beta2      = (const float*)d_in[7];
    const float* Wq         = (const float*)d_in[8];
    const float* bq         = (const float*)d_in[9];
    const float* Wk         = (const float*)d_in[10];
    const float* bk         = (const float*)d_in[11];
    const float* Wv         = (const float*)d_in[12];
    const float* bv         = (const float*)d_in[13];
    const int*   nbr        = (const int*)d_in[14];
    const int*   cond       = (const int*)d_in[15];
    float* out = (float*)d_out;

    // Workspace partition (all chunk sizes multiples of 256 bytes)
    char* ws = (char*)d_ws;
    float*  tmp   = (float*)ws;            ws += (size_t)NVOX * CCH * 4;   // 76.8 MB
    bf16_t* featB = (bf16_t*)ws;           ws += (size_t)NVOX * CCH * 2;   // 38.4 MB
    bf16_t* x1B   = (bf16_t*)ws;           ws += (size_t)NVOX * CCH * 2;
    bf16_t* outB  = (bf16_t*)ws;           ws += (size_t)NVOX * CCH * 2;
    bf16_t* W1t   = (bf16_t*)ws;           ws += (size_t)KOFF * CCH * CCH * 2;
    bf16_t* W2t   = (bf16_t*)ws;           ws += (size_t)KOFF * CCH * CCH * 2;
    bf16_t* WkT   = (bf16_t*)ws;           ws += (size_t)CCH * CCH * 2;
    bf16_t* qB    = (bf16_t*)ws;           ws += (size_t)PBANK * CCH * 2;
    bf16_t* vT    = (bf16_t*)ws;           ws += (size_t)CCH * PBANK * 2;
    float*  stats = (float*)ws;            // 4*96 floats: [sum1,sq1,sum2,sq2]
    float*  stats1 = stats;
    float*  stats2 = stats + 2 * CCH;

    const int totalFeat = NVOX * CCH;
    const int totalW    = KOFF * CCH * CCH;

    // 0) zero BN stats
    zero_stats_kernel<<<1, 384, 0, stream>>>(stats);

    // 1) preps
    prep_w_kernel<<<(totalW + 255) / 256, 256, 0, stream>>>(W1, W1t, totalW);
    prep_w_kernel<<<(totalW + 255) / 256, 256, 0, stream>>>(W2, W2t, totalW);
    prep_wk_kernel<<<(CCH * CCH + 255) / 256, 256, 0, stream>>>(Wk, WkT);
    prep_feat_kernel<<<(totalFeat + 255) / 256, 256, 0, stream>>>(voxel_feat, featB, totalFeat);
    prep_qv_kernel<<<(PBANK * CCH + 255) / 256, 256, 0, stream>>>(point_feat, Wq, bq, Wv, bv, qB, vT);

    // 2) conv1 -> tmp   (32 rows/wave * 8 waves = 256 rows per block)
    const int convBlocks = (NVOX + 255) / 256;    // 782
    conv_wmma_kernel<<<convBlocks, 256, 0, stream>>>(featB, W1t, nbr, tmp, NVOX);

    // 3) BN1 stats + apply (+ReLU) -> x1B
    bn_stats_kernel<<<200, 96, 0, stream>>>(tmp, stats1, NVOX, 1000);
    bn1_apply_kernel<<<(totalFeat + 255) / 256, 256, 0, stream>>>(tmp, stats1, gamma1, beta1, cond, x1B, totalFeat);

    // 4) conv2 -> tmp (reused)
    conv_wmma_kernel<<<convBlocks, 256, 0, stream>>>(x1B, W2t, nbr, tmp, NVOX);

    // 5) BN2 stats + apply + residual + ReLU -> out[:, :96] and outB (bf16)
    bn_stats_kernel<<<200, 96, 0, stream>>>(tmp, stats2, NVOX, 1000);
    bn2_apply_kernel<<<(totalFeat + 255) / 256, 256, 0, stream>>>(tmp, stats2, gamma2, beta2, cond,
                                                                  voxel_feat, out, outB, totalFeat);

    // 6) fused cross-attention + LayerNorm -> out[:, 96:192]
    const int attnBlocks = NVOX / 64;             // 3125 (exact)
    attn_wmma_kernel<<<attnBlocks, 128, 0, stream>>>(outB, WkT, bk, qB, vT, out, NVOX);
}
